// LiquidNeuralNetwork_17575006175781
// MI455X (gfx1250) — matrix operand order repaired and checked by reading, not compiled, not run on hardware
//
#include <hip/hip_runtime.h>
#include <hip/hip_bf16.h>

// ---------------------------------------------------------------------------
// Liquid NN (LTC, RK4) persistent-scan kernel for gfx1250 (MI455X).
//
// pack_weights: one-shot repack of all GEMM B-operands into d_ws as bf16 in
//   WMMA-fragment-contiguous layout (each wave-fragment = 1KB block).
// ltc_scan: 32 WGs x 512 threads (16 wave32s, 4 waves/SIMD). Each WG owns 16
//   batch rows, runs the whole 512-step scan privately. Each wave owns one
//   16-column tile and computes BOTH rec (hh@Wrec.T) and gate (hh@Wg_h.T)
//   results; the RK4 elementwise update runs IN REGISTERS on the wave's own
//   tile (xin/xg/ksum/hh-f32/h-f32 are per-lane register arrays). Only the
//   recurrence-essential state crosses waves through LDS: hhb16 (next stage's
//   WMMA A source) and the stage-3 pre-LN values for LayerNorm.
//   One barrier per RK4 stage. B-fragments stream from packed L2 panels.
// ---------------------------------------------------------------------------

typedef __attribute__((ext_vector_type(16))) __bf16 v16bf;
typedef __attribute__((ext_vector_type(8)))  float  v8f;

union FragBF { v16bf v; uint4 q[2]; };

#define B_   512
#define T_   512
#define FS_  64
#define FC_  32
#define H_   256
#define HB_  16
#define NTHREADS 512

// ---- d_ws packed-weight panel offsets (elements of bf16) -------------------
#define WS_REC0 0          // Wrec0            16 tiles x 8 kf -> 65536
#define WS_GH0  65536      // Wg0[:, 96:352]                   -> 65536
#define WS_REC1 131072     // Wrec1                            -> 65536
#define WS_GH1  196608     // Wg1[:, 256:512]                  -> 65536
#define WS_IN0  262144     // Win0             16 tiles x 3 kf -> 24576
#define WS_GX0  286720     // Wg0[:, 0:96]                     -> 24576
#define WS_IN1  311296     // Win1                             -> 65536
#define WS_GX1  376832     // Wg1[:, 0:256]                    -> 65536
#define WS_ELEMS 442368    // * 2 bytes = 884736 B of d_ws used

// ---- LDS layout (bytes) ----------------------------------------------------
#define OFF_H32    0        // float [2][16][256]  persistent layer states
#define OFF_HB16   32768    // bf16  [2][16][256]  bf16 copies (WMMA A source)
#define OFF_HH32   49152    // float [16][256]     stage-3 pre-LN values (also head hid buf)
#define OFF_HHB16  65536    // bf16  [16][256]     current RK4 state (WMMA A source)
#define OFF_XT     73728    // bf16  [16][96]      layer-0 input tile
#define OFF_ITAU   76800    // float [2][256]      1/(softplus(tau)+1)
#define OFF_LNG    78848    // float [2][256]
#define OFF_LNB    80896    // float [2][256]
#define OFF_BG     82944    // float [2][256]
#define OFF_STAT   84992    // float [32]          mu[16], rstd[16]
#define SMEM_BYTES 85120

// Load a pre-packed bf16 fragment (32 contiguous bytes per lane) from d_ws.
__device__ __forceinline__ v16bf load_frag_ws(const __bf16* p) {
  FragBF f;
  const uint4* q = (const uint4*)p;
  f.q[0] = q[0];
  f.q[1] = q[1];
  return f.v;
}

// A-fragment (16x32 bf16, M=row, K) from row-major LDS bf16 [16][rowstride].
__device__ __forceinline__ v16bf load_afrag_lds(const __bf16* base, int rowstride,
                                                int kb, int lane) {
  int m  = lane & 15;
  int lh = lane >> 4;
  int c0 = kb * 32 + lh * 8;
  FragBF f;
  f.q[0] = *(const uint4*)(base + m * rowstride + c0);
  f.q[1] = *(const uint4*)(base + m * rowstride + c0 + 16);
  return f.v;
}

__device__ __forceinline__ v8f wmma_bf16(v16bf a, v16bf b, v8f c) {
  return __builtin_amdgcn_wmma_f32_16x16x32_bf16(
      false, a, false, b, (short)0, c, false, false);
}

// ---------------------------------------------------------------------------
// Packing kernel: grid (256, 8), block 256. Panel p handled by blockIdx.y.
// ---------------------------------------------------------------------------
extern "C" __global__ void pack_weights(const float* __restrict__ Wrec0,
                                        const float* __restrict__ Wg0,
                                        const float* __restrict__ Win0,
                                        const float* __restrict__ Wrec1,
                                        const float* __restrict__ Wg1,
                                        const float* __restrict__ Win1,
                                        __bf16* __restrict__ ws) {
  const int p = blockIdx.y;
  const int i = blockIdx.x * blockDim.x + threadIdx.x;
  const int nkf_t[8]  = {8, 8, 8, 8, 3, 3, 8, 8};
  const int rowl_t[8] = {256, 352, 256, 512, 96, 352, 256, 512};
  const int coff_t[8] = {0, 96, 0, 256, 0, 0, 0, 0};
  const int dst_t[8]  = {WS_REC0, WS_GH0, WS_REC1, WS_GH1,
                         WS_IN0,  WS_GX0, WS_IN1,  WS_GX1};
  const int nel_t[8]  = {65536, 65536, 65536, 65536, 24576, 24576, 65536, 65536};
  if (i >= nel_t[p]) return;
  const float* src = (p == 0) ? Wrec0 : (p == 1) ? Wg0 : (p == 2) ? Wrec1
                   : (p == 3) ? Wg1   : (p == 4) ? Win0 : (p == 5) ? Wg0
                   : (p == 6) ? Win1  : Wg1;
  int e  = i & 15;
  int ln = (i >> 4) & 31;
  int fk = i >> 9;
  int kf = fk % nkf_t[p];
  int w  = fk / nkf_t[p];
  int n  = w * 16 + (ln & 15);
  int k  = kf * 32 + (e >> 3) * 16 + (ln >> 4) * 8 + (e & 7);
  ws[dst_t[p] + i] = (__bf16)src[(size_t)n * rowl_t[p] + coff_t[p] + k];
}

// ---------------------------------------------------------------------------
// Main persistent scan kernel.
// ---------------------------------------------------------------------------
extern "C" __global__ __launch_bounds__(NTHREADS, 1)
void ltc_scan_kernel(const float* __restrict__ seq,  const float* __restrict__ ctx,
                     const float* __restrict__ tau0, const float* __restrict__ bg0,
                     const float* __restrict__ lng0, const float* __restrict__ lnb0,
                     const float* __restrict__ tau1, const float* __restrict__ bg1,
                     const float* __restrict__ lng1, const float* __restrict__ lnb1,
                     const float* __restrict__ cW1,  const float* __restrict__ cb1,
                     const float* __restrict__ cW2,  const float* __restrict__ cb2,
                     const __bf16* __restrict__ ws,  float* __restrict__ out) {
  extern __shared__ char smem[];
  float*  h32   = (float*)(smem + OFF_H32);
  __bf16* hb16  = (__bf16*)(smem + OFF_HB16);
  float*  hh32  = (float*)(smem + OFF_HH32);
  __bf16* hhb16 = (__bf16*)(smem + OFF_HHB16);
  __bf16* xtb   = (__bf16*)(smem + OFF_XT);
  float*  itau  = (float*)(smem + OFF_ITAU);
  float*  lngs  = (float*)(smem + OFF_LNG);
  float*  lnbs  = (float*)(smem + OFF_LNB);
  float*  bgs   = (float*)(smem + OFF_BG);
  float*  stat  = (float*)(smem + OFF_STAT);

  const int tid   = threadIdx.x;
  const int lane  = tid & 31;
  const int wave  = tid >> 5;            // 0..15, one N-column tile pair each
  const int wg    = blockIdx.x;          // 0..31, owns batch rows wg*16..+15
  const int ncol  = wave * 16 + (lane & 15);
  const int khalf = lane >> 4;

  // ---- one-time init -------------------------------------------------------
#pragma unroll 2
  for (int it = 0; it < (2 * HB_ * H_) / NTHREADS; ++it) {
    int id = tid + it * NTHREADS;
    h32[id] = 0.f; hb16[id] = (__bf16)0.f;
  }
  {
    int id = tid;                        // 2*H_ == NTHREADS == 512
    int l = id >> 8, j = id & 255;
    float x  = (l ? tau1 : tau0)[j];
    float sp = (x > 20.f) ? x : log1pf(expf(x));   // softplus
    itau[id] = 1.f / (sp + 1.f);
    lngs[id] = (l ? lng1 : lng0)[j];
    lnbs[id] = (l ? lnb1 : lnb0)[j];
    bgs[id]  = (l ? bg1  : bg0 )[j];
  }
  __syncthreads();

  // ---- time scan -----------------------------------------------------------
#pragma unroll 1
  for (int t = 0; t < T_; ++t) {
    // layer-0 input tile: [seq(64) | ctx(32)] -> bf16 [16][96]
#pragma unroll
    for (int it = 0; it < (HB_ * 96) / NTHREADS; ++it) {
      int id = tid + it * NTHREADS;
      int m = id / 96, f = id - m * 96;
      int b = wg * HB_ + m;
      float v = (f < FS_) ? seq[((size_t)b * T_ + t) * FS_ + f]
                          : ctx[(size_t)b * FC_ + (f - FS_)];
      xtb[id] = (__bf16)v;
    }
    __syncthreads();

#pragma unroll
    for (int l = 0; l < 2; ++l) {
      const int kfin = l ? 8 : 3;
      float*  h32l  = h32  + l * HB_ * H_;
      __bf16* hb16l = hb16 + l * HB_ * H_;

      // Per-wave streamed fragment bases (lane-resolved, this wave's tile).
      const __bf16* wbR = ws + (l ? WS_REC1 : WS_REC0) + wave * (8 * 512) + lane * 16;
      const __bf16* wbG = ws + (l ? WS_GH1  : WS_GH0 ) + wave * (8 * 512) + lane * 16;
      const __bf16* wbI = ws + (l ? WS_IN1  : WS_IN0 ) + wave * (kfin * 512) + lane * 16;
      const __bf16* wbX = ws + (l ? WS_GX1  : WS_GX0 ) + wave * (kfin * 512) + lane * 16;

      // Per-lane column constants.
      const float itau_c = itau[l * H_ + ncol];
      const float lng_c  = lngs[l * H_ + ncol];
      const float lnb_c  = lnbs[l * H_ + ncol];
      const float badd   = bgs [l * H_ + ncol];

      // Projections kept in registers: xinr = x@Win.T, xgr = x@Wg_x.T + bg.
      float xinr[8], xgr[8];
      {
        const __bf16* Asrc = l ? hb16 : xtb;     // layer1 input = new h0
        const int ars = l ? H_ : 96;
        v8f accR, accG;
#pragma unroll
        for (int r = 0; r < 8; ++r) { accR[r] = 0.f; accG[r] = 0.f; }
#pragma unroll
        for (int kf = 0; kf < kfin; ++kf) {
          v16bf a  = load_afrag_lds(Asrc, ars, kf, lane);
          v16bf bI = load_frag_ws(wbI + kf * 512);
          v16bf bX = load_frag_ws(wbX + kf * 512);
          accR = wmma_bf16(a, bI, accR);
          accG = wmma_bf16(a, bX, accG);
        }
#pragma unroll
        for (int r = 0; r < 8; ++r) { xinr[r] = accR[r]; xgr[r] = accG[r] + badd; }
      }

      // Per-lane RK4 state in registers (this wave's own 8 elements/lane).
      float hreg[8], hhreg[8], ksr[8], hnreg[8];
#pragma unroll
      for (int r = 0; r < 8; ++r) {
        hreg[r]  = h32l[(r + 8 * khalf) * H_ + ncol];
        hhreg[r] = hreg[r];
        ksr[r]   = 0.f;
        hnreg[r] = 0.f;
      }

      // ---- RK4 stages: GEMM + in-register elementwise, ONE barrier each ----
#pragma unroll
      for (int s = 0; s < 4; ++s) {
        const __bf16* Ah = (s == 0) ? hb16l : hhb16;
        v8f accR, accG;
#pragma unroll
        for (int r = 0; r < 8; ++r) { accR[r] = 0.f; accG[r] = 0.f; }
#pragma unroll
        for (int kf = 0; kf < 8; ++kf) {
          v16bf a  = load_afrag_lds(Ah, H_, kf, lane);
          v16bf bR = load_frag_ws(wbR + kf * 512);  // streamed from L2
          v16bf bG = load_frag_ws(wbG + kf * 512);  // streamed from L2
          accR = wmma_bf16(a, bR, accR);
          accG = wmma_bf16(a, bG, accG);
        }
        const float kw = (s == 0 || s == 3) ? (1.f / 6.f) : (2.f / 6.f);
        const float aw = (s == 2) ? 1.0f : 0.5f;
#pragma unroll
        for (int r = 0; r < 8; ++r) {
          float gp = accG[r] + xgr[r];
          float g  = 1.f / (1.f + expf(-tanhf(gp)));
          float k  = -hhreg[r] * itau_c + xinr[r] + g * accR[r];
          ksr[r]   = (s == 0) ? kw * k : ksr[r] + kw * k;
          if (s < 3) {
            hhreg[r] = hreg[r] + aw * k;
            hhb16[(r + 8 * khalf) * H_ + ncol] = (__bf16)hhreg[r];
          } else {
            hnreg[r] = hreg[r] + ksr[r];                  // pre-LN value
            hh32[(r + 8 * khalf) * H_ + ncol] = hnreg[r]; // for LN stats
          }
        }
        __syncthreads();
      }

      // ---- LayerNorm stats: one wave per batch row (16 waves, 16 rows) ----
      {
        int m = wave;
        float s1 = 0.f, s2 = 0.f;
#pragma unroll
        for (int i = 0; i < 8; ++i) {
          float v = hh32[m * H_ + lane + 32 * i];
          s1 += v; s2 += v * v;
        }
#pragma unroll
        for (int d = 16; d >= 1; d >>= 1) {
          s1 += __shfl_xor(s1, d);
          s2 += __shfl_xor(s2, d);
        }
        if (lane == 0) {
          float mu  = s1 * (1.f / H_);
          float var = s2 * (1.f / H_) - mu * mu;
          stat[m]      = mu;
          stat[16 + m] = rsqrtf(var + 1e-5f);
        }
      }
      __syncthreads();

      // apply LN, tanh, clamp on own tile -> new h_l (f32 + bf16)
#pragma unroll
      for (int r = 0; r < 8; ++r) {
        int m = r + 8 * khalf;
        float hn = (hnreg[r] - stat[m]) * stat[16 + m] * lng_c + lnb_c;
        float v = tanhf(hn);
        v = fminf(10.f, fmaxf(-10.f, v));
        h32l[m * H_ + ncol]  = v;
        hb16l[m * H_ + ncol] = (__bf16)v;
      }
      __syncthreads();
    } // layers
  } // time

  // ---- readout head: out = relu(h1 @ cW1.T + cb1) @ cW2.T + cb2 ------------
  // hid buffer [16][128] reuses hh32.
#pragma unroll
  for (int it = 0; it < (HB_ * 128) / NTHREADS; ++it) {
    int id = tid + it * NTHREADS;
    int m = id >> 7, j = id & 127;
    const float* hr = h32 + HB_ * H_ + m * H_;   // layer-1 state
    const float* w  = cW1 + j * H_;
    float s = 0.f;
    for (int q = 0; q < H_; ++q) s += hr[q] * w[q];
    s += cb1[j];
    hh32[m * 128 + j] = fmaxf(s, 0.f);
  }
  __syncthreads();
  {
    int m = wave;                         // 16 waves, 16 rows
    float s = 0.f;
#pragma unroll
    for (int i = 0; i < 4; ++i)
      s += hh32[m * 128 + lane + 32 * i] * cW2[lane + 32 * i];
#pragma unroll
    for (int d = 16; d >= 1; d >>= 1) s += __shfl_xor(s, d);
    if (lane == 0) out[wg * HB_ + m] = s + cb2[0];
  }
}

extern "C" void kernel_launch(void* const* d_in, const int* in_sizes, int n_in,
                              void* d_out, int out_size, void* d_ws, size_t ws_size,
                              hipStream_t stream) {
  (void)in_sizes; (void)n_in; (void)ws_size; (void)out_size;
  const float* seq   = (const float*)d_in[0];
  const float* ctx   = (const float*)d_in[1];
  const float* tau0  = (const float*)d_in[2];
  const float* Win0  = (const float*)d_in[3];
  const float* Wrec0 = (const float*)d_in[4];
  const float* Wg0   = (const float*)d_in[5];
  const float* bg0   = (const float*)d_in[6];
  const float* lng0  = (const float*)d_in[7];
  const float* lnb0  = (const float*)d_in[8];
  const float* tau1  = (const float*)d_in[9];
  const float* Win1  = (const float*)d_in[10];
  const float* Wrec1 = (const float*)d_in[11];
  const float* Wg1   = (const float*)d_in[12];
  const float* bg1   = (const float*)d_in[13];
  const float* lng1  = (const float*)d_in[14];
  const float* lnb1  = (const float*)d_in[15];
  const float* cW1   = (const float*)d_in[16];
  const float* cb1   = (const float*)d_in[17];
  const float* cW2   = (const float*)d_in[18];
  const float* cb2   = (const float*)d_in[19];
  float* out  = (float*)d_out;
  __bf16* wsb = (__bf16*)d_ws;   // needs >= 884736 bytes

  // 1) repack weights into bf16 WMMA fragments (deterministic every call).
  pack_weights<<<dim3(256, 8), dim3(256), 0, stream>>>(
      Wrec0, Wg0, Win0, Wrec1, Wg1, Win1, wsb);

  // 2) persistent scan.
  ltc_scan_kernel<<<dim3(B_ / HB_), dim3(NTHREADS), SMEM_BYTES, stream>>>(
      seq, ctx,
      tau0, bg0, lng0, lnb0,
      tau1, bg1, lng1, lnb1,
      cW1, cb1, cW2, cb2, wsb, out);
}
